// Length_Regulator_15857019257479
// MI455X (gfx1250) — matrix operand order repaired and checked by reading, not compiled, not run on hardware
//
#include <hip/hip_runtime.h>

#define D_MODEL 384
#define LSEQ    512
#define BSZ     32
#define LP      (LSEQ + 2)      // padded sequence length (zero halo at 0 and LP-1)
#define LN_EPS  1e-5f

typedef __bf16 bf16;
typedef __attribute__((ext_vector_type(16))) __bf16 v16bf;
typedef __attribute__((ext_vector_type(8)))  __bf16 v8bf;
typedef __attribute__((ext_vector_type(8)))  float  v8f;

__device__ __forceinline__ bf16 f2bf(float f) {
  unsigned u = __builtin_bit_cast(unsigned, f);
  unsigned r = u + 0x7FFFu + ((u >> 16) & 1u);   // round-to-nearest-even
  unsigned short h = (unsigned short)(r >> 16);
  return __builtin_bit_cast(bf16, h);
}

// ---- fp32 weights [co][ci][k] -> bf16 [k][co][ci] ----------------------------
__global__ __launch_bounds__(256) void cvt_w_k(const float* __restrict__ w,
                                               bf16* __restrict__ wb) {
  int o = blockIdx.x * 256 + threadIdx.x;            // 3*384*384 total, exact
  int k  = o / (D_MODEL * D_MODEL);
  int r  = o - k * (D_MODEL * D_MODEL);
  int co = r / D_MODEL;
  int ci = r - co * D_MODEL;
  wb[o] = f2bf(w[(co * D_MODEL + ci) * 3 + k]);
}

// ---- x fp32 [B][L][D] -> bf16 padded [B][LP][D], halos zeroed ----------------
__global__ __launch_bounds__(256) void cvt_x_k(const float* __restrict__ x,
                                               bf16* __restrict__ xb) {
  int o = blockIdx.x * 256 + threadIdx.x;            // B*LP*D total, exact
  int row = o / D_MODEL;
  int c   = o - row * D_MODEL;
  int b   = row / LP;
  int lp  = row - b * LP;
  float v = 0.f;
  if (lp > 0 && lp < LP - 1)
    v = x[((size_t)b * LSEQ + (lp - 1)) * D_MODEL + c];
  xb[o] = f2bf(v);
}

// ---- zero the halo rows of a padded bf16 buffer ------------------------------
__global__ __launch_bounds__(256) void zero_halo_k(bf16* __restrict__ buf) {
  int o = blockIdx.x * 256 + threadIdx.x;            // B*2*D total, exact
  int b = o / (2 * D_MODEL);
  int r = o - b * (2 * D_MODEL);
  int lp = (r < D_MODEL) ? 0 : (LP - 1);
  int c  = r % D_MODEL;
  buf[((size_t)b * LP + lp) * D_MODEL + c] = f2bf(0.f);
}

// ---- Conv1d(k=3,SAME)+ReLU as 3 accumulated bf16 WMMA GEMMs ------------------
// wave computes a 16(co) x 64(l) strip; tasks = B * 24 * 8 = 6144; 8 waves/block
template<bool OUT_PADDED_BF16>
__global__ __launch_bounds__(256) void conv_wmma_k(
    const bf16* __restrict__ xin,    // [B][LP][D] padded
    const bf16* __restrict__ wbf,    // [3][D][D]
    const float* __restrict__ bias,  // [D]
    bf16*  __restrict__ out_bf,      // [B][LP][D] (interior rows only)
    float* __restrict__ out_f)       // [B][L][D]
{
  const int lane = threadIdx.x & 31;
  const int hi   = lane >> 4;
  const int lnm  = lane & 15;
  const int wid  = blockIdx.x * 8 + (threadIdx.x >> 5);
  const int b       = wid / (24 * 8);
  const int rem     = wid - b * (24 * 8);
  const int co_tile = rem >> 3;
  const int l_strip = rem & 7;
  const int co_base = co_tile * 16;
  const int l_base  = l_strip * 64;

  v8f acc0 = {}, acc1 = {}, acc2 = {}, acc3 = {};
  const int arow = co_base + lnm;
  const size_t xbatch = (size_t)b * LP * D_MODEL;

  #pragma unroll
  for (int k = 0; k < 3; ++k) {
    const bf16* wrow = wbf + (size_t)(k * D_MODEL + arow) * D_MODEL;
    // output l consumes xpad row (l + k); lane's B column is l_base+j*16+lnm
    const bf16* xr0 = xin + xbatch + (size_t)(l_base + lnm + k) * D_MODEL;
    const bf16* xr1 = xr0 + (size_t)16 * D_MODEL;
    const bf16* xr2 = xr0 + (size_t)32 * D_MODEL;
    const bf16* xr3 = xr0 + (size_t)48 * D_MODEL;
    #pragma unroll
    for (int kk = 0; kk < 12; ++kk) {
      const int ciA = kk * 32 + hi * 16;
      v8bf alo = *(const v8bf*)(wrow + ciA);
      v8bf ahi = *(const v8bf*)(wrow + ciA + 8);
      v16bf afrag = __builtin_shufflevector(
          alo, ahi, 0, 1, 2, 3, 4, 5, 6, 7, 8, 9, 10, 11, 12, 13, 14, 15);
      const int ciB = kk * 32 + hi * 16;
      v16bf b0 = *(const v16bf*)(xr0 + ciB);
      v16bf b1 = *(const v16bf*)(xr1 + ciB);
      v16bf b2 = *(const v16bf*)(xr2 + ciB);
      v16bf b3 = *(const v16bf*)(xr3 + ciB);
      acc0 = __builtin_amdgcn_wmma_f32_16x16x32_bf16(false, afrag, false, b0,
                                                     (short)0, acc0, false, false);
      acc1 = __builtin_amdgcn_wmma_f32_16x16x32_bf16(false, afrag, false, b1,
                                                     (short)0, acc1, false, false);
      acc2 = __builtin_amdgcn_wmma_f32_16x16x32_bf16(false, afrag, false, b2,
                                                     (short)0, acc2, false, false);
      acc3 = __builtin_amdgcn_wmma_f32_16x16x32_bf16(false, afrag, false, b3,
                                                     (short)0, acc3, false, false);
    }
  }

  const v8f bias8 = *(const v8f*)(bias + co_base + hi * 8);
  v8f accs[4] = {acc0, acc1, acc2, acc3};
  #pragma unroll
  for (int j = 0; j < 4; ++j) {
    const int l = l_base + j * 16 + lnm;
    if constexpr (OUT_PADDED_BF16) {
      v8bf st;
      #pragma unroll
      for (int v = 0; v < 8; ++v)
        st[v] = f2bf(fmaxf(accs[j][v] + bias8[v], 0.f));
      *(v8bf*)(out_bf + xbatch + (size_t)(l + 1) * D_MODEL + co_base + hi * 8) = st;
    } else {
      v8f st;
      #pragma unroll
      for (int v = 0; v < 8; ++v)
        st[v] = fmaxf(accs[j][v] + bias8[v], 0.f);
      *(v8f*)(out_f + ((size_t)b * LSEQ + l) * D_MODEL + co_base + hi * 8) = st;
    }
  }
}

// ---- LayerNorm over D=384; one block per (b,l) row ---------------------------
template<bool TO_BF16_PAD>
__global__ __launch_bounds__(128) void layernorm_k(
    const float* __restrict__ in, const float* __restrict__ g,
    const float* __restrict__ beta, bf16* __restrict__ out_bf,
    float* __restrict__ out_f)
{
  __shared__ float s1[128], s2[128];
  const int row = blockIdx.x;
  const int tid = threadIdx.x;
  const float* xr = in + (size_t)row * D_MODEL;
  float v0 = xr[tid], v1 = xr[tid + 128], v2 = xr[tid + 256];
  s1[tid] = v0 + v1 + v2;
  s2[tid] = v0 * v0 + v1 * v1 + v2 * v2;
  __syncthreads();
  for (int off = 64; off > 0; off >>= 1) {
    if (tid < off) { s1[tid] += s1[tid + off]; s2[tid] += s2[tid + off]; }
    __syncthreads();
  }
  const float mu   = s1[0] * (1.f / D_MODEL);
  const float var  = s2[0] * (1.f / D_MODEL) - mu * mu;
  const float rstd = rsqrtf(var + LN_EPS);
  const int b = row / LSEQ, l = row - b * LSEQ;
  #pragma unroll
  for (int i = 0; i < 3; ++i) {
    const int c = tid + i * 128;
    const float v = (i == 0) ? v0 : (i == 1) ? v1 : v2;
    const float y = (v - mu) * rstd * g[c] + beta[c];
    if constexpr (TO_BF16_PAD)
      out_bf[((size_t)b * LP + l + 1) * D_MODEL + c] = f2bf(y);
    else
      out_f[(size_t)row * D_MODEL + c] = y;
  }
}

// ---- final D->1 linear: one wave per row, shuffle reduction ------------------
__global__ __launch_bounds__(256) void linear_k(
    const float* __restrict__ h, const float* __restrict__ w,
    const float* __restrict__ bsc, float* __restrict__ dur)
{
  const int lane = threadIdx.x & 31;
  const int row  = blockIdx.x * 8 + (threadIdx.x >> 5);
  const float* xr = h + (size_t)row * D_MODEL;
  float s = 0.f;
  #pragma unroll
  for (int i = 0; i < 12; ++i) { int c = lane + i * 32; s += xr[c] * w[c]; }
  #pragma unroll
  for (int off = 16; off > 0; off >>= 1) s += __shfl_xor(s, off, 32);
  if (lane == 0) dur[row] = s + bsc[0];
}

// ---- inclusive scan of durations per batch (Hillis-Steele, 512 threads) ------
__global__ __launch_bounds__(512) void scan_k(const int* __restrict__ dur,
                                              int* __restrict__ cum) {
  __shared__ int s[512];
  const int b = blockIdx.x, tid = threadIdx.x;
  s[tid] = dur[b * LSEQ + tid];
  __syncthreads();
  for (int off = 1; off < 512; off <<= 1) {
    int v = (tid >= off) ? s[tid - off] : 0;
    __syncthreads();
    s[tid] += v;
    __syncthreads();
  }
  cum[b * LSEQ + tid] = s[tid];
}

// ---- length regulator: out[b,t,:] = x[b, searchsorted(cum,t,'right'), :] -----
__global__ __launch_bounds__(96) void gather_k(const float* __restrict__ x,
                                               const int* __restrict__ cum,
                                               float* __restrict__ out, int T) {
  const int b = blockIdx.y, t = blockIdx.x, tid = threadIdx.x;
  const int* cb = cum + b * LSEQ;
  const int total = cb[LSEQ - 1];
  float4* orow = (float4*)(out + ((size_t)b * T + t) * D_MODEL);
  if (t >= total) { orow[tid] = make_float4(0.f, 0.f, 0.f, 0.f); return; }
  int lo = 0, hiq = LSEQ;
  while (lo < hiq) { int mid = (lo + hiq) >> 1; if (cb[mid] > t) hiq = mid; else lo = mid + 1; }
  const float4* xrow = (const float4*)(x + ((size_t)b * LSEQ + lo) * D_MODEL);
  orow[tid] = xrow[tid];
}

extern "C" void kernel_launch(void* const* d_in, const int* in_sizes, int n_in,
                              void* d_out, int out_size, void* d_ws, size_t ws_size,
                              hipStream_t stream) {
  (void)in_sizes; (void)n_in; (void)ws_size;
  const float* x     = (const float*)d_in[0];
  const int*   durs  = (const int*)  d_in[1];
  const float* c1a_w = (const float*)d_in[2];
  const float* c1a_b = (const float*)d_in[3];
  const float* c1b_w = (const float*)d_in[4];
  const float* c1b_b = (const float*)d_in[5];
  const float* ln1_g = (const float*)d_in[6];
  const float* ln1_b = (const float*)d_in[7];
  const float* c2a_w = (const float*)d_in[8];
  const float* c2a_b = (const float*)d_in[9];
  const float* c2b_w = (const float*)d_in[10];
  const float* c2b_b = (const float*)d_in[11];
  const float* ln2_g = (const float*)d_in[12];
  const float* ln2_b = (const float*)d_in[13];
  const float* lin_w = (const float*)d_in[14];
  const float* lin_b = (const float*)d_in[15];
  float* out = (float*)d_out;

  const int T = (out_size - BSZ * LSEQ) / (BSZ * D_MODEL);

  char* ws = (char*)d_ws;
  size_t off = 0;
  auto alloc = [&](size_t bytes) {
    void* p = ws + off;
    off += (bytes + 255) & ~(size_t)255;
    return p;
  };
  bf16*  bufA = (bf16*) alloc(sizeof(bf16) * (size_t)BSZ * LP * D_MODEL);
  bf16*  bufB = (bf16*) alloc(sizeof(bf16) * (size_t)BSZ * LP * D_MODEL);
  float* bufF = (float*)alloc(sizeof(float) * (size_t)BSZ * LSEQ * D_MODEL);
  bf16*  w1a  = (bf16*) alloc(sizeof(bf16) * 3 * D_MODEL * D_MODEL);
  bf16*  w1b  = (bf16*) alloc(sizeof(bf16) * 3 * D_MODEL * D_MODEL);
  bf16*  w2a  = (bf16*) alloc(sizeof(bf16) * 3 * D_MODEL * D_MODEL);
  bf16*  w2b  = (bf16*) alloc(sizeof(bf16) * 3 * D_MODEL * D_MODEL);
  int*   cum  = (int*)  alloc(sizeof(int) * BSZ * LSEQ);

  const int WGRID = (3 * D_MODEL * D_MODEL) / 256;   // 1728
  cvt_w_k<<<WGRID, 256, 0, stream>>>(c1a_w, w1a);
  cvt_w_k<<<WGRID, 256, 0, stream>>>(c1b_w, w1b);
  cvt_w_k<<<WGRID, 256, 0, stream>>>(c2a_w, w2a);
  cvt_w_k<<<WGRID, 256, 0, stream>>>(c2b_w, w2b);
  cvt_x_k<<<(BSZ * LP * D_MODEL) / 256, 256, 0, stream>>>(x, bufA);   // 24672
  zero_halo_k<<<(BSZ * 2 * D_MODEL) / 256, 256, 0, stream>>>(bufB);   // 96

  const int CONV_GRID = (BSZ * 24 * 8) / 8;  // 768 blocks of 8 waves
  conv_wmma_k<true ><<<CONV_GRID, 256, 0, stream>>>(bufA, w1a, c1a_b, bufB, nullptr);
  conv_wmma_k<false><<<CONV_GRID, 256, 0, stream>>>(bufB, w1b, c1b_b, nullptr, bufF);
  layernorm_k<true ><<<BSZ * LSEQ, 128, 0, stream>>>(bufF, ln1_g, ln1_b, bufB, nullptr);
  conv_wmma_k<true ><<<CONV_GRID, 256, 0, stream>>>(bufB, w2a, c2a_b, bufA, nullptr);
  conv_wmma_k<false><<<CONV_GRID, 256, 0, stream>>>(bufA, w2b, c2b_b, nullptr, bufF);
  layernorm_k<false><<<BSZ * LSEQ, 128, 0, stream>>>(bufF, ln2_g, ln2_b, nullptr, bufF);

  linear_k<<<(BSZ * LSEQ) / 8, 256, 0, stream>>>(
      bufF, lin_w, lin_b, out + (size_t)BSZ * T * D_MODEL);

  scan_k<<<BSZ, 512, 0, stream>>>(durs, cum);
  if (T > 0) {
    dim3 g(T, BSZ);
    gather_k<<<g, 96, 0, stream>>>(x, cum, out, T);
  }
}
